// Mamba2Block_78357383348460
// MI455X (gfx1250) — compile-verified, matrix operand order checked by reading, not run on hardware
//
#include <hip/hip_runtime.h>
#include <hip/hip_bf16.h>

// ---------------------------------------------------------------------------
// Mamba2 block for MI455X (gfx1250, wave32, WMMA).
// GEMMs: v_wmma_f32_16x16x32_bf16, double-buffered LDS staged via
// GLOBAL_LOAD_ASYNC_TO_LDS_B128 (ASYNCcnt). Block tile 128x128, wave tile
// 32x64 (2x4 WMMA accumulators) for high WMMA-per-LDS-read intensity.
// Weights are pre-transposed/padded so all tile staging is branch-free 16B.
// ---------------------------------------------------------------------------

typedef __attribute__((ext_vector_type(16))) __bf16 v16bf;
typedef __attribute__((ext_vector_type(8)))  __bf16 v8bf;
typedef __attribute__((ext_vector_type(8)))  float  v8f;
typedef int v4i_vec __attribute__((vector_size(4 * sizeof(int))));

#define D_MODEL 1024
#define L_SEQ   2048
#define BATCH   2
#define NTOK    (BATCH * L_SEQ)          // 4096 tokens
#define H_HEADS 16
#define N_STATE 16
#define K_CONV  4
#define D_INNER 2048
#define P_HEAD  128
#define NPROJ   4624                      // 2*2048 + 2*256 + 16
#define NPROJ_PAD 4736                    // ceil(4624/128)*128
#define OFF_Z   2048
#define OFF_B   4096
#define OFF_C   4352
#define OFF_DT  4608

#if defined(__has_builtin)
#if __has_builtin(__builtin_amdgcn_global_load_async_to_lds_b128)
#define HAVE_ASYNC_LDS 1
#endif
#endif

// 16-byte global -> LDS copy; async DMA when available.
__device__ __forceinline__ void copy16_to_lds(__bf16* lds, const __bf16* g) {
#if defined(HAVE_ASYNC_LDS)
  __builtin_amdgcn_global_load_async_to_lds_b128(
      (__attribute__((address_space(1))) v4i_vec*)(void*)g,
      (__attribute__((address_space(3))) v4i_vec*)(void*)lds, 0, 0);
#else
  *(v8bf*)lds = *(const v8bf*)g;
#endif
}

__device__ __forceinline__ void wait_async_lds() {
#if defined(HAVE_ASYNC_LDS)
#if __has_builtin(__builtin_amdgcn_s_wait_asynccnt)
  __builtin_amdgcn_s_wait_asynccnt(0);
#else
  asm volatile("s_wait_asynccnt 0x0" ::: "memory");
#endif
#endif
}

// ---------------------------------------------------------------------------
// W[K,N] f32 -> Wt[Npad,K] bf16 (transposed, zero-padded rows N..Npad)
// ---------------------------------------------------------------------------
__global__ void mamba_transpose_cast_bf16(const float* __restrict__ W,
                                          __bf16* __restrict__ Wt, int K,
                                          int N, int Npad) {
  int idx = blockIdx.x * blockDim.x + threadIdx.x;
  if (idx >= Npad * K) return;
  int n = idx / K;
  int k = idx - n * K;
  Wt[idx] = (n < N) ? (__bf16)W[(size_t)k * N + n] : (__bf16)0.0f;
}

// ---------------------------------------------------------------------------
// LayerNorm over D=1024, one block per token, emit bf16
// ---------------------------------------------------------------------------
__global__ __launch_bounds__(256) void mamba_ln_bf16(
    const float* __restrict__ x, const float* __restrict__ gamma,
    const float* __restrict__ beta, __bf16* __restrict__ xn) {
  __shared__ float red[256];
  const int tok = blockIdx.x;
  const int t = threadIdx.x;
  const float* row = x + (size_t)tok * D_MODEL;

  float v[4];
  float s = 0.f;
#pragma unroll
  for (int i = 0; i < 4; ++i) { v[i] = row[t * 4 + i]; s += v[i]; }
  red[t] = s; __syncthreads();
  for (int o = 128; o > 0; o >>= 1) { if (t < o) red[t] += red[t + o]; __syncthreads(); }
  const float mu = red[0] * (1.0f / D_MODEL);
  __syncthreads();

  float sq = 0.f;
#pragma unroll
  for (int i = 0; i < 4; ++i) { float d = v[i] - mu; sq += d * d; }
  red[t] = sq; __syncthreads();
  for (int o = 128; o > 0; o >>= 1) { if (t < o) red[t] += red[t + o]; __syncthreads(); }
  const float rstd = rsqrtf(red[0] * (1.0f / D_MODEL) + 1e-6f);

  __bf16* orow = xn + (size_t)tok * D_MODEL;
#pragma unroll
  for (int i = 0; i < 4; ++i) {
    int c = t * 4 + i;
    orow[c] = (__bf16)((v[i] - mu) * rstd * gamma[c] + beta[c]);
  }
}

// ---------------------------------------------------------------------------
// WMMA GEMM: C[M,N](f32) = A[M,K](bf16,row) * Bt[Npad,K](bf16, B transposed)
// Block tile 128x128, 8 waves (4 M x 2 N), wave tile 32x64 = 2x4 WMMA accs.
// Double-buffered LDS ([rows][40] bf16 padded stride), async staged.
// Fragment per lane: two ds_read_b128, K = (e%8)+8*(lane/16)+16*(e/8).
// ---------------------------------------------------------------------------
template <bool ADD_RES>
__global__ __launch_bounds__(256) void mamba_gemm_wmma(
    const __bf16* __restrict__ A, const __bf16* __restrict__ Bt,
    float* __restrict__ C, const float* __restrict__ R, int M, int N,
    int Kd) {
  constexpr int AS = 40;  // padded LDS row stride, keeps 16B alignment
  constexpr int BS = 40;
  __shared__ __align__(16) __bf16 lA[2][128 * AS];
  __shared__ __align__(16) __bf16 lB[2][128 * BS];

  const int t = threadIdx.x;
  const int lane = t & 31;
  const int wave = t >> 5;
  const int wm = wave & 3;       // 4 waves along M (32 rows each)
  const int wn = wave >> 2;      // 2 waves along N (64 cols each)
  const int rb = blockIdx.y * 128;
  const int cb = blockIdx.x * 128;
  const int lr = lane & 15;
  const int kg = (lane >> 4) * 8;     // K sub-group per half-wave
  const int rhalf = (lane >> 4) * 8;  // C/D row offset per half-wave

  // staging coordinates: 128 rows x 32 cols = 512 16B-chunks each for A and B
  const int r0 = (t + 0) >> 2, c0 = ((t + 0) & 3) * 8;
  const int r1 = (t + 256) >> 2, c1 = ((t + 256) & 3) * 8;

  auto stage = [&](int buf, int k0) {
    copy16_to_lds(&lA[buf][r0 * AS + c0], A + (size_t)(rb + r0) * Kd + k0 + c0);
    copy16_to_lds(&lA[buf][r1 * AS + c1], A + (size_t)(rb + r1) * Kd + k0 + c1);
    copy16_to_lds(&lB[buf][r0 * BS + c0], Bt + (size_t)(cb + r0) * Kd + k0 + c0);
    copy16_to_lds(&lB[buf][r1 * BS + c1], Bt + (size_t)(cb + r1) * Kd + k0 + c1);
  };

  v8f acc[2][4] = {};
  int buf = 0;
  stage(0, 0);

  for (int k0 = 0; k0 < Kd; k0 += 32) {
    wait_async_lds();
    __syncthreads();  // staged tile `buf` visible to all waves
    if (k0 + 32 < Kd) stage(buf ^ 1, k0 + 32);

    v16bf afrag[2], bfrag[4];
#pragma unroll
    for (int tm = 0; tm < 2; ++tm) {
      const __bf16* base = &lA[buf][(wm * 32 + tm * 16 + lr) * AS];
      v8bf lo = *(const v8bf*)(base + kg);
      v8bf hi = *(const v8bf*)(base + 16 + kg);
      afrag[tm] = __builtin_shufflevector(lo, hi, 0, 1, 2, 3, 4, 5, 6, 7, 8, 9,
                                          10, 11, 12, 13, 14, 15);
    }
#pragma unroll
    for (int tn = 0; tn < 4; ++tn) {
      const __bf16* base = &lB[buf][(wn * 64 + tn * 16 + lr) * BS];
      v8bf lo = *(const v8bf*)(base + kg);
      v8bf hi = *(const v8bf*)(base + 16 + kg);
      bfrag[tn] = __builtin_shufflevector(lo, hi, 0, 1, 2, 3, 4, 5, 6, 7, 8, 9,
                                          10, 11, 12, 13, 14, 15);
    }
#pragma unroll
    for (int tm = 0; tm < 2; ++tm)
#pragma unroll
      for (int tn = 0; tn < 4; ++tn)
        acc[tm][tn] = __builtin_amdgcn_wmma_f32_16x16x32_bf16(
            false, afrag[tm], false, bfrag[tn], (short)0, acc[tm][tn], false,
            false);
    buf ^= 1;
  }

  // ---- epilogue: C/D layout -> lane = N col, VGPR v = row (v | v+8) ----
#pragma unroll
  for (int tm = 0; tm < 2; ++tm)
#pragma unroll
    for (int tn = 0; tn < 4; ++tn) {
      int col = cb + wn * 64 + tn * 16 + lr;
      if (col < N) {
#pragma unroll
        for (int v = 0; v < 8; ++v) {
          int row = rb + wm * 32 + tm * 16 + rhalf + v;
          size_t idx = (size_t)row * N + col;
          float val = acc[tm][tn][v];
          C[idx] = ADD_RES ? (val + R[idx]) : val;
        }
      }
    }
}

// ---------------------------------------------------------------------------
// Fused: causal depthwise conv(K=4) + SiLU on x-branch; softplus dt * mask.
// ---------------------------------------------------------------------------
__global__ __launch_bounds__(256) void mamba_conv_dt(
    const float* __restrict__ proj, const float* __restrict__ conv_w,
    const float* __restrict__ conv_b, const float* __restrict__ wmask,
    float* __restrict__ xc, float* __restrict__ dtws) {
  const int tok = blockIdx.x;
  const int l = tok & (L_SEQ - 1);
  const int t = threadIdx.x;
  const float* prow = proj + (size_t)tok * NPROJ;

  if (t < H_HEADS) {
    float raw = prow[OFF_DT + t];
    float sp = (raw > 20.f) ? raw : log1pf(__expf(raw));
    dtws[tok * H_HEADS + t] = sp * (1.0f - wmask[tok]);
  }

  const int c0 = t * 8;
  float acc[8];
#pragma unroll
  for (int j = 0; j < 8; ++j) acc[j] = conv_b[c0 + j];
#pragma unroll
  for (int k = 0; k < K_CONV; ++k) {
    int ls = l - (K_CONV - 1) + k;
    if (ls >= 0) {
      const float* xrow = proj + (size_t)(tok - (K_CONV - 1) + k) * NPROJ;
      const float* w = conv_w + k * D_INNER + c0;
#pragma unroll
      for (int j = 0; j < 8; ++j) acc[j] = fmaf(xrow[c0 + j], w[j], acc[j]);
    }
  }
  float* orow = xc + (size_t)tok * D_INNER;
#pragma unroll
  for (int j = 0; j < 8; ++j) {
    float a = acc[j];
    orow[c0 + j] = a / (1.0f + __expf(-a));  // SiLU
  }
}

// ---------------------------------------------------------------------------
// Sequential selective scan. One workgroup per (b,h); 128 lanes own one
// p-row each, 16 states in registers. z-gate SiLU fused, emits bf16 y.
// ---------------------------------------------------------------------------
__global__ __launch_bounds__(128) void mamba_scan(
    const float* __restrict__ proj, const float* __restrict__ xc,
    const float* __restrict__ dtws, const float* __restrict__ A_log,
    __bf16* __restrict__ ybf) {
  const int b = blockIdx.x >> 4;
  const int h = blockIdx.x & 15;
  const int p = threadIdx.x;

  float a[N_STATE], st[N_STATE];
#pragma unroll
  for (int n = 0; n < N_STATE; ++n) {
    a[n] = -__expf(A_log[h * N_STATE + n]);
    st[n] = 0.f;
  }

  for (int l = 0; l < L_SEQ; ++l) {
    const size_t tok = (size_t)b * L_SEQ + l;
    const float* prow = proj + tok * NPROJ;
    const float dt = dtws[tok * H_HEADS + h];
    const float xv = xc[tok * D_INNER + h * P_HEAD + p];
    const float dtx = dt * xv;
    float y = 0.f;
#pragma unroll
    for (int n = 0; n < N_STATE; ++n) {
      float dec = __expf(dt * a[n]);
      float Bv = prow[OFF_B + h * N_STATE + n];
      float Cv = prow[OFF_C + h * N_STATE + n];
      st[n] = fmaf(dec, st[n], dtx * Bv);
      y = fmaf(st[n], Cv, y);
    }
    float z = prow[OFF_Z + h * P_HEAD + p];
    float g = z / (1.0f + __expf(-z));
    ybf[tok * D_INNER + h * P_HEAD + p] = (__bf16)(y * g);
  }
}

// ---------------------------------------------------------------------------
// Launch
// ---------------------------------------------------------------------------
extern "C" void kernel_launch(void* const* d_in, const int* in_sizes, int n_in,
                              void* d_out, int out_size, void* d_ws,
                              size_t ws_size, hipStream_t stream) {
  (void)in_sizes; (void)n_in; (void)out_size; (void)ws_size;
  const float* x      = (const float*)d_in[0];
  const float* wmask  = (const float*)d_in[1];
  const float* ln_s   = (const float*)d_in[2];
  const float* ln_b   = (const float*)d_in[3];
  const float* W_in   = (const float*)d_in[4];
  const float* conv_w = (const float*)d_in[5];
  const float* conv_b = (const float*)d_in[6];
  const float* A_log  = (const float*)d_in[7];
  const float* W_out  = (const float*)d_in[8];
  float* out = (float*)d_out;

  char* ws = (char*)d_ws;
  size_t off = 0;
  auto suballoc = [&](size_t bytes) -> char* {
    char* p = ws + off;
    off += (bytes + 255) & ~(size_t)255;
    return p;
  };
  __bf16* xn   = (__bf16*)suballoc((size_t)NTOK * D_MODEL * 2);
  __bf16* WiT  = (__bf16*)suballoc((size_t)NPROJ_PAD * D_MODEL * 2);
  float*  proj = (float*) suballoc((size_t)NTOK * NPROJ * 4);
  float*  xc   = (float*) suballoc((size_t)NTOK * D_INNER * 4);
  float*  dtws = (float*) suballoc((size_t)NTOK * H_HEADS * 4);
  __bf16* ybf  = (__bf16*)suballoc((size_t)NTOK * D_INNER * 2);
  __bf16* WoT  = (__bf16*)suballoc((size_t)D_MODEL * D_INNER * 2);

  // weight transpose+cast+pad (deterministic, redone every launch)
  mamba_transpose_cast_bf16<<<(NPROJ_PAD * D_MODEL + 255) / 256, 256, 0,
                              stream>>>(W_in, WiT, D_MODEL, NPROJ, NPROJ_PAD);
  mamba_transpose_cast_bf16<<<(D_MODEL * D_INNER + 255) / 256, 256, 0,
                              stream>>>(W_out, WoT, D_INNER, D_MODEL, D_MODEL);

  // LN -> bf16
  mamba_ln_bf16<<<NTOK, 256, 0, stream>>>(x, ln_s, ln_b, xn);

  // proj = xn @ W_in  (M=4096, N=4624, K=1024)
  mamba_gemm_wmma<false><<<dim3(NPROJ_PAD / 128, NTOK / 128), 256, 0,
                           stream>>>(xn, WiT, proj, nullptr, NTOK, NPROJ,
                                     D_MODEL);

  // conv + SiLU + dt
  mamba_conv_dt<<<NTOK, 256, 0, stream>>>(proj, conv_w, conv_b, wmask, xc,
                                          dtws);

  // selective scan + z-gate
  mamba_scan<<<BATCH * H_HEADS, 128, 0, stream>>>(proj, xc, dtws, A_log, ybf);

  // out = y @ W_out + residual  (M=4096, N=1024, K=2048)
  mamba_gemm_wmma<true><<<dim3(D_MODEL / 128, NTOK / 128), 256, 0, stream>>>(
      ybf, WoT, out, x, NTOK, D_MODEL, D_INNER);
}
